// Attention_23270132809979
// MI455X (gfx1250) — compile-verified, matrix operand order checked by reading, not run on hardware
//
#include <hip/hip_runtime.h>
#include <hip/hip_bf16.h>

// ---------------------------------------------------------------------------
// Types for CDNA5 WMMA
// ---------------------------------------------------------------------------
typedef __attribute__((ext_vector_type(16))) __bf16 v16bf;
typedef __attribute__((ext_vector_type(8)))  float  v8f;

// Problem constants (match the JAX reference)
#define BATCH   4
#define SEQ     2048
#define HEADS   16
#define DHEAD   64
#define DIMM    1024
#define MROWS   (BATCH * SEQ)      // 8192

// ---------------------------------------------------------------------------
// Helpers
// ---------------------------------------------------------------------------
static __device__ __forceinline__ unsigned short f32_to_bf16(float f) {
    unsigned u = __builtin_bit_cast(unsigned, f);
    unsigned r = u + 0x7fffu + ((u >> 16) & 1u);   // round-to-nearest-even
    return (unsigned short)(r >> 16);
}

// gfx1250 async copy: global -> LDS, 16B per lane, tracked by ASYNCcnt.
// saddr form: mem_addr = SGPR_base + VGPR_offset(32b); LDS addr from VGPR.
static __device__ __forceinline__ void async_ld128(const void* sbase,
                                                   unsigned goff_bytes,
                                                   unsigned lds_bytes) {
    asm volatile("global_load_async_to_lds_b128 %0, %1, %2"
                 :
                 : "v"(lds_bytes), "v"(goff_bytes), "s"(sbase)
                 : "memory");
}

template <int N>
static __device__ __forceinline__ void s_wait_asynccnt() {
    asm volatile("s_wait_asynccnt %0" ::"n"(N) : "memory");
}

// LDS byte offset of a __shared__ object: flat LDS aperture keeps the
// workgroup-relative byte offset in addr[31:0] (ISA 10.2 aperture table).
static __device__ __forceinline__ unsigned lds_off(const void* p) {
    return (unsigned)(size_t)p;
}

// Build a 16-bit WMMA fragment (A or B) from LDS.
// ISA layout (16-bit 16x32 A): lanes 0-15 K-base 0, lanes 16-31 K-base 8;
// VGPR v<4 holds K = base+2v, base+2v+1; VGPR v>=4 holds K = 16+base+2(v-4).
// B (32x16) mirrors this with column = lane%16 and `stride` between K rows.
static __device__ __forceinline__ v16bf load_frag(const unsigned short* p, int stride) {
    const int lane = threadIdx.x & 31;
    const int kb = (lane & 16) ? 8 : 0;
    v16bf f;
#pragma unroll
    for (int v = 0; v < 8; ++v) {
        int k = (v < 4) ? (kb + 2 * v) : (16 + kb + 2 * (v - 4));
        f[2 * v]     = __builtin_bit_cast(__bf16, p[(k)     * stride]);
        f[2 * v + 1] = __builtin_bit_cast(__bf16, p[(k + 1) * stride]);
    }
    return f;
}

static __device__ __forceinline__ v8f wmma_bf16(v16bf a, v16bf b, v8f c) {
    return __builtin_amdgcn_wmma_f32_16x16x32_bf16(
        false, a, false, b, (short)0, c, false, false);
}

// ---------------------------------------------------------------------------
// f32 -> bf16 conversion kernel
// ---------------------------------------------------------------------------
__global__ void cvt_bf16_kernel(const float* __restrict__ src,
                                unsigned short* __restrict__ dst, int n) {
    int i = blockIdx.x * blockDim.x + threadIdx.x;
    int step = gridDim.x * blockDim.x;
    for (; i < n; i += step) dst[i] = f32_to_bf16(src[i]);
}

// ---------------------------------------------------------------------------
// Tiled bf16 GEMM:  C[M, NN] = A[M,K] * W[K,NN] + bias
//   EPI = 0 : write f32 row-major
//   EPI = 1 : write bf16 scattered to [b,h,n,d] (QKV layout)
// Block tile 128x64, BK=64, double-buffered async LDS staging.
// 256 threads = 8 waves; wave (waveM, waveN) owns a 32x32 strip (2x2 WMMA).
// ---------------------------------------------------------------------------
template <int EPI>
__global__ __launch_bounds__(256) void gemm_bf16_kernel(
    const unsigned short* __restrict__ A,
    const unsigned short* __restrict__ W,
    const float* __restrict__ bias,
    void* __restrict__ outp)
{
    constexpr int K = DIMM, NN = DIMM;
    constexpr int BM = 128, BN = 64, BK = 64;
    constexpr int KT = K / BK;  // 16

    __shared__ unsigned short As[2][BM * BK];  // 2 x 16KB
    __shared__ unsigned short Bs[2][BK * BN];  // 2 x 8KB

    const int tid   = threadIdx.x;
    const int lane  = tid & 31;
    const int wave  = tid >> 5;
    const int waveM = wave & 3;
    const int waveN = wave >> 2;
    const int bm    = blockIdx.y;
    const int bn    = blockIdx.x;
    const int half8 = (lane & 16) ? 8 : 0;

    const unsigned asb = lds_off(&As[0][0]);
    const unsigned bsb = lds_off(&Bs[0][0]);

    // Per stage: A = 1024 x 16B chunks (4 instr/wave), B = 512 (2 instr/wave)
    auto issue_tiles = [&](int kt, int st) {
#pragma unroll
        for (int i = 0; i < 4; ++i) {
            int c = tid + i * 256;
            int row = c >> 3, seg = c & 7;  // A row length = 64 ush = 8 chunks
            unsigned go = (unsigned)((((bm * BM + row) * K) + kt * BK + seg * 8) * 2);
            async_ld128(A, go, asb + (unsigned)((st * BM * BK + row * BK + seg * 8) * 2));
        }
#pragma unroll
        for (int i = 0; i < 2; ++i) {
            int c = tid + i * 256;
            int row = c >> 3, seg = c & 7;  // B row length = 64 ush = 8 chunks
            unsigned go = (unsigned)((((kt * BK + row) * NN) + bn * BN + seg * 8) * 2);
            async_ld128(W, go, bsb + (unsigned)((st * BK * BN + row * BN + seg * 8) * 2));
        }
    };

    v8f acc[2][2] = {};

    issue_tiles(0, 0);
    for (int kt = 0; kt < KT; ++kt) {
        const int st = kt & 1;
        if (kt + 1 < KT) {
            issue_tiles(kt + 1, st ^ 1);
            s_wait_asynccnt<6>();   // only next-stage copies may remain
        } else {
            s_wait_asynccnt<0>();
        }
        __syncthreads();

        const unsigned short* Asl = &As[st][0];
        const unsigned short* Bsl = &Bs[st][0];
#pragma unroll
        for (int ks = 0; ks < BK; ks += 32) {
            v16bf afr[2], bfr[2];
#pragma unroll
            for (int mi = 0; mi < 2; ++mi)
                afr[mi] = load_frag(Asl + (waveM * 32 + mi * 16 + (lane & 15)) * BK + ks, 1);
#pragma unroll
            for (int ni = 0; ni < 2; ++ni)
                bfr[ni] = load_frag(Bsl + ks * BN + waveN * 32 + ni * 16 + (lane & 15), BN);
#pragma unroll
            for (int mi = 0; mi < 2; ++mi)
#pragma unroll
                for (int ni = 0; ni < 2; ++ni)
                    acc[mi][ni] = wmma_bf16(afr[mi], bfr[ni], acc[mi][ni]);
        }
        __syncthreads();
    }

    // Epilogue: C layout -> lane l, VGPR r: row = r + half8, col = lane%16
#pragma unroll
    for (int mi = 0; mi < 2; ++mi) {
#pragma unroll
        for (int ni = 0; ni < 2; ++ni) {
            const int col = bn * BN + waveN * 32 + ni * 16 + (lane & 15);
            const float bv = bias[col];
#pragma unroll
            for (int r = 0; r < 8; ++r) {
                const int m = bm * BM + waveM * 32 + mi * 16 + r + half8;
                const float val = acc[mi][ni][r] + bv;
                if (EPI == 0) {
                    reinterpret_cast<float*>(outp)[(size_t)m * NN + col] = val;
                } else {
                    const int h = col >> 6, d = col & 63;
                    const int b = m >> 11, n = m & (SEQ - 1);
                    reinterpret_cast<unsigned short*>(outp)
                        [((size_t)(b * HEADS + h) * SEQ + n) * DHEAD + d] = f32_to_bf16(val);
                }
            }
        }
    }
}

// ---------------------------------------------------------------------------
// Flash attention: one block = 128 queries of one (b,h); 256 threads = 8
// waves, each wave owns a 16-row strip. Online softmax; K/V tiles double-
// buffered via async-to-LDS; P staged through per-wave LDS for re-fragmenting.
// ---------------------------------------------------------------------------
__global__ __launch_bounds__(256) void attn_kernel(
    const unsigned short* __restrict__ Q,
    const unsigned short* __restrict__ Km,
    const unsigned short* __restrict__ Vm,
    unsigned short* __restrict__ ctx)
{
    __shared__ unsigned short Qs[128 * 64];     // 16KB
    __shared__ unsigned short Ks[2][64 * 64];   // 16KB
    __shared__ unsigned short Vs[2][64 * 64];   // 16KB
    __shared__ unsigned short Ps[8 * 16 * 64];  // 16KB per-wave P staging

    const int tid   = threadIdx.x;
    const int lane  = tid & 31;
    const int wave  = tid >> 5;
    const int bh    = blockIdx.y;   // b*HEADS + h
    const int qt    = blockIdx.x;   // 128-query tile
    const int half8 = (lane & 16) ? 8 : 0;

    const unsigned short* Qh = Q  + (size_t)bh * SEQ * DHEAD;
    const unsigned short* Kh = Km + (size_t)bh * SEQ * DHEAD;
    const unsigned short* Vh = Vm + (size_t)bh * SEQ * DHEAD;

    const unsigned qsb = lds_off(&Qs[0]);
    const unsigned ksb = lds_off(&Ks[0][0]);
    const unsigned vsb = lds_off(&Vs[0][0]);

    // Q tile: 128x64 ush = 1024 chunks -> 4 async instr/wave
#pragma unroll
    for (int i = 0; i < 4; ++i) {
        int c = tid + i * 256;
        int row = c >> 3, seg = c & 7;
        unsigned go = (unsigned)((((qt * 128 + row) * DHEAD) + seg * 8) * 2);
        async_ld128(Qh, go, qsb + (unsigned)((row * 64 + seg * 8) * 2));
    }
    // K,V tiles: 64x64 ush each = 512 chunks -> 2+2 async instr/wave per stage
    auto issue_kv = [&](int kt, int st) {
#pragma unroll
        for (int i = 0; i < 2; ++i) {
            int c = tid + i * 256;
            int row = c >> 3, seg = c & 7;
            unsigned go = (unsigned)((((kt * 64 + row) * DHEAD) + seg * 8) * 2);
            unsigned lo = (unsigned)((st * 4096 + row * 64 + seg * 8) * 2);
            async_ld128(Kh, go, ksb + lo);
            async_ld128(Vh, go, vsb + lo);
        }
    };

    v16bf qfr0, qfr1;
    float mrow[8], lrow[8];
#pragma unroll
    for (int r = 0; r < 8; ++r) { mrow[r] = -1e30f; lrow[r] = 0.f; }
    v8f o[4] = {};

    issue_kv(0, 0);
    for (int kt = 0; kt < SEQ / 64; ++kt) {
        const int st = kt & 1;
        if (kt + 1 < SEQ / 64) {
            issue_kv(kt + 1, st ^ 1);
            s_wait_asynccnt<4>();   // Q + stage kt are complete
        } else {
            s_wait_asynccnt<0>();
        }
        __syncthreads();

        if (kt == 0) {
            qfr0 = load_frag(&Qs[(wave * 16 + (lane & 15)) * 64 + 0], 1);
            qfr1 = load_frag(&Qs[(wave * 16 + (lane & 15)) * 64 + 32], 1);
        }

        const unsigned short* Ksl = &Ks[st][0];
        const unsigned short* Vsl = &Vs[st][0];

        // S strip (16 x 64) = Q strip (16 x 64) * K tile^T
        v8f s[4] = {};
#pragma unroll
        for (int j = 0; j < 4; ++j) {
            v16bf b0 = load_frag(Ksl + (j * 16 + (lane & 15)) * 64 + 0, 1);
            v16bf b1 = load_frag(Ksl + (j * 16 + (lane & 15)) * 64 + 32, 1);
            s[j] = wmma_bf16(qfr0, b0, s[j]);
            s[j] = wmma_bf16(qfr1, b1, s[j]);
        }

        // Online softmax (scale = 1/sqrt(64) = 0.125)
#pragma unroll
        for (int r = 0; r < 8; ++r) {
            float rm = fmaxf(fmaxf(s[0][r], s[1][r]), fmaxf(s[2][r], s[3][r])) * 0.125f;
#pragma unroll
            for (int msk = 1; msk <= 8; msk <<= 1)
                rm = fmaxf(rm, __shfl_xor(rm, msk, 32));
            const float mn = fmaxf(mrow[r], rm);
            const float alpha = __expf(mrow[r] - mn);
            mrow[r] = mn;
            float psum = 0.f;
#pragma unroll
            for (int j = 0; j < 4; ++j) {
                const float p = __expf(s[j][r] * 0.125f - mn);
                psum += p;
                Ps[wave * 1024 + (r + half8) * 64 + j * 16 + (lane & 15)] = f32_to_bf16(p);
            }
#pragma unroll
            for (int msk = 1; msk <= 8; msk <<= 1)
                psum += __shfl_xor(psum, msk, 32);
            lrow[r] = lrow[r] * alpha + psum;
#pragma unroll
            for (int j = 0; j < 4; ++j) o[j][r] *= alpha;
        }

        // Our own LDS writes to Ps must land before re-reading as fragments
        asm volatile("s_wait_dscnt 0x0" ::: "memory");

        // O strip (16 x 64) += P strip (16 x 64) * V tile (64 x 64)
        v16bf pa0 = load_frag(&Ps[wave * 1024 + (lane & 15) * 64 + 0], 1);
        v16bf pa1 = load_frag(&Ps[wave * 1024 + (lane & 15) * 64 + 32], 1);
#pragma unroll
        for (int j = 0; j < 4; ++j) {
            v16bf vb0 = load_frag(Vsl + 0 * 64 + j * 16 + (lane & 15), 64);
            v16bf vb1 = load_frag(Vsl + 32 * 64 + j * 16 + (lane & 15), 64);
            o[j] = wmma_bf16(pa0, vb0, o[j]);
            o[j] = wmma_bf16(pa1, vb1, o[j]);
        }
        __syncthreads();
    }

    // Epilogue: ctx[b, n, h*64 + d] as bf16 for the final projection
    const int b = bh >> 4, h = bh & 15;
#pragma unroll
    for (int j = 0; j < 4; ++j) {
#pragma unroll
        for (int r = 0; r < 8; ++r) {
            const int qrow = qt * 128 + wave * 16 + r + half8;
            const float val = o[j][r] / lrow[r];
            ctx[((size_t)(b * SEQ + qrow)) * DIMM + h * 64 + j * 16 + (lane & 15)] =
                f32_to_bf16(val);
        }
    }
}

// ---------------------------------------------------------------------------
// Host-side launch
// ---------------------------------------------------------------------------
extern "C" void kernel_launch(void* const* d_in, const int* in_sizes, int n_in,
                              void* d_out, int out_size, void* d_ws, size_t ws_size,
                              hipStream_t stream) {
    (void)in_sizes; (void)n_in; (void)out_size; (void)ws_size;

    const float* x  = (const float*)d_in[0];
    const float* Wq = (const float*)d_in[1];
    const float* bq = (const float*)d_in[2];
    const float* Wk = (const float*)d_in[3];
    const float* bk = (const float*)d_in[4];
    const float* Wv = (const float*)d_in[5];
    const float* bv = (const float*)d_in[6];
    const float* Wo = (const float*)d_in[7];
    const float* bo = (const float*)d_in[8];

    char* ws = (char*)d_ws;
    size_t off = 0;
    auto alloc = [&](size_t bytes) -> void* {
        void* p = ws + off;
        off += (bytes + 255) & ~(size_t)255;
        return p;
    };
    const size_t NX = (size_t)MROWS * DIMM;   // 8192*1024
    const size_t NW = (size_t)DIMM * DIMM;    // 1024*1024

    unsigned short* Xb   = (unsigned short*)alloc(NX * 2);
    unsigned short* Wqb  = (unsigned short*)alloc(NW * 2);
    unsigned short* Wkb  = (unsigned short*)alloc(NW * 2);
    unsigned short* Wvb  = (unsigned short*)alloc(NW * 2);
    unsigned short* Wob  = (unsigned short*)alloc(NW * 2);
    unsigned short* Qb   = (unsigned short*)alloc(NX * 2);
    unsigned short* Kb   = (unsigned short*)alloc(NX * 2);
    unsigned short* Vb   = (unsigned short*)alloc(NX * 2);
    unsigned short* Ctxb = (unsigned short*)alloc(NX * 2);

    // 1) f32 -> bf16 conversions
    cvt_bf16_kernel<<<2048, 256, 0, stream>>>(x,  Xb,  (int)NX);
    cvt_bf16_kernel<<<1024, 256, 0, stream>>>(Wq, Wqb, (int)NW);
    cvt_bf16_kernel<<<1024, 256, 0, stream>>>(Wk, Wkb, (int)NW);
    cvt_bf16_kernel<<<1024, 256, 0, stream>>>(Wv, Wvb, (int)NW);
    cvt_bf16_kernel<<<1024, 256, 0, stream>>>(Wo, Wob, (int)NW);

    // 2) QKV projections (bias fused, scattered to [b,h,n,d] bf16)
    dim3 ggrid(DIMM / 64, MROWS / 128);  // (16, 64)
    gemm_bf16_kernel<1><<<ggrid, 256, 0, stream>>>(Xb, Wqb, bq, Qb);
    gemm_bf16_kernel<1><<<ggrid, 256, 0, stream>>>(Xb, Wkb, bk, Kb);
    gemm_bf16_kernel<1><<<ggrid, 256, 0, stream>>>(Xb, Wvb, bv, Vb);

    // 3) Flash attention -> ctx (bf16, [b, n, h*d])
    dim3 agrid(SEQ / 128, BATCH * HEADS);  // (16, 64)
    attn_kernel<<<agrid, 256, 0, stream>>>(Qb, Kb, Vb, Ctxb);

    // 4) Output projection -> f32 out
    gemm_bf16_kernel<0><<<ggrid, 256, 0, stream>>>(Ctxb, Wob, bo, (float*)d_out);
}
